// HyperLSTMCell_90409061580836
// MI455X (gfx1250) — matrix operand order repaired and checked by reading, not compiled, not run on hardware
//
#include <hip/hip_runtime.h>
#include <cstdint>
#include <cstddef>

#define BB    8192
#define EMBD  512
#define HIDD  1024
#define HYPD  256
#define NZD   64

typedef __attribute__((ext_vector_type(16))) __bf16 v16bf;
typedef __attribute__((ext_vector_type(8)))  __bf16 v8bf;
typedef __attribute__((ext_vector_type(8)))  float  v8f;

// ---------------- WMMA fragment helpers (wave32, 16x16x32 bf16) ----------------
// A tile staged in LDS as [M][32] row-major (bf16). Per ISA layout, lane L
// (row = L&15, half = L>>4) needs k = half*8 + {0..7} and 16 + half*8 + {0..7}:
// two contiguous 16B reads.
__device__ __forceinline__ v16bf frag_A(const __bf16* base) {
  const int lane = threadIdx.x & 31;
  const __bf16* r = base + (lane & 15) * 32 + (lane >> 4) * 8;
  v8bf lo = *(const v8bf*)(r);
  v8bf hi = *(const v8bf*)(r + 16);
  v16bf a;
#pragma unroll
  for (int i = 0; i < 8; ++i) { a[i] = lo[i]; a[8 + i] = hi[i]; }
  return a;
}

// B tile staged in LDS transposed as [N][32] (bf16): lane L (col = L&15,
// half = L>>4) reads k = half*8 + {0..7} and 16 + half*8 + {0..7}: two 16B reads.
__device__ __forceinline__ v16bf frag_B(const __bf16* base) {
  const int lane = threadIdx.x & 31;
  const __bf16* r = base + (lane & 15) * 32 + (lane >> 4) * 8;
  v8bf lo = *(const v8bf*)(r);
  v8bf hi = *(const v8bf*)(r + 16);
  v16bf b;
#pragma unroll
  for (int i = 0; i < 8; ++i) { b[i] = lo[i]; b[8 + i] = hi[i]; }
  return b;
}

__device__ __forceinline__ v8f wmma_bf16(v16bf a, v16bf b, v8f c) {
  return __builtin_amdgcn_wmma_f32_16x16x32_bf16(false, a, false, b,
                                                 (short)0, c, false, false);
}

__device__ __forceinline__ void zero_acc(v8f* a, int n) {
  for (int t = 0; t < n; ++t)
#pragma unroll
    for (int v = 0; v < 8; ++v) a[t][v] = 0.0f;
}

__device__ __forceinline__ float sigf(float x) { return 1.0f / (1.0f + __expf(-x)); }

// ---------------- K0: Wc[g] = dW[g]^T @ w[g]  (64 x N, K = 1024) ----------------
// dW: [4][HID][NZ] f32, w: [4][HID][N] f32. Output WcT: [4][N][NZ] bf16 (N-major).
__global__ void __launch_bounds__(256)
k_combine(const float* __restrict__ dW, const float* __restrict__ wM,
          __bf16* __restrict__ WcT, int N) {
  const int nb = blockIdx.x, g = blockIdx.y;
  const int tid = threadIdx.x;
  const int wid = tid >> 5, lane = tid & 31;
  const int wm = wid & 3, wn = wid >> 2;
  const int hl = lane >> 4, cl = lane & 15;

  __shared__ __align__(16) __bf16 sA[64 * 32];
  __shared__ __align__(16) __bf16 sBT[128 * 32];

  const float* Ag = dW + (size_t)g * HIDD * NZD;
  const float* Bg = wM + (size_t)g * HIDD * N + (size_t)nb * 128;

  v8f acc[4]; zero_acc(acc, 4);

  for (int k0 = 0; k0 < HIDD; k0 += 32) {
    __syncthreads();
    { // sA[m][c] = dW[g][k0+c][m]  (coalesced across threads in m)
      const int m = tid & 63, c0 = (tid >> 6) * 8;
#pragma unroll
      for (int c = 0; c < 8; ++c)
        sA[m * 32 + c0 + c] = (__bf16)Ag[(size_t)(k0 + c0 + c) * NZD + m];
    }
    { // sBT[n][c] = w[g][k0+c][nb*128+n] (coalesced across threads in n)
      const int n = tid & 127, c0 = (tid >> 7) * 16;
#pragma unroll
      for (int c = 0; c < 16; ++c)
        sBT[n * 32 + c0 + c] = (__bf16)Bg[(size_t)(k0 + c0 + c) * N + n];
    }
    __syncthreads();
    v16bf a = frag_A(sA + wm * 16 * 32);
#pragma unroll
    for (int t = 0; t < 4; ++t) {
      v16bf b = frag_B(sBT + (wn * 64 + t * 16) * 32);
      acc[t] = wmma_bf16(a, b, acc[t]);
    }
  }

  __bf16* out = WcT + (size_t)g * N * NZD;
#pragma unroll
  for (int t = 0; t < 4; ++t) {
    const int ncol = nb * 128 + wn * 64 + t * 16 + cl;
#pragma unroll
    for (int v = 0; v < 8; ++v) {
      const int m = wm * 16 + hl * 8 + v;
      out[(size_t)ncol * NZD + m] = (__bf16)acc[t][v];
    }
  }
}

// ------- K1: ifgo = e@Wx^T + h_hat@Wh^T + b, per-gate LayerNorm (fused) -------
// Block tile 128 x 256 (= one full gate). Each of the 8 waves owns 16 complete
// rows of the gate in 16 accumulators, so the LayerNorm is register-resident:
// intra-lane sum over accumulators + shfl_xor across the 16-lane half. The
// WMMA C layout guarantees each lane only needs stats for rows it holds.
__global__ void __launch_bounds__(256)
k_hyper_ifgo(const float* __restrict__ e, const float* __restrict__ hhat,
             const float* __restrict__ Wx, const float* __restrict__ Wh,
             const float* __restrict__ bias,
             const float* __restrict__ lng, const float* __restrict__ lnb,
             float* __restrict__ ifgo_ln) {
  const int mb = blockIdx.x * 128, g = blockIdx.y;
  const int tid = threadIdx.x;
  const int wid = tid >> 5, lane = tid & 31;
  const int hl = lane >> 4, cl = lane & 15;

  __shared__ __align__(16) __bf16 sA[128 * 32];   //  8 KB
  __shared__ __align__(16) __bf16 sBT[256 * 32];  // 16 KB

  v8f acc[16]; zero_acc(acc, 16);

  for (int ph = 0; ph < 2; ++ph) {
    const float* A = ph ? hhat : e;
    const float* W = ph ? Wh : Wx;
    const int K = ph ? HYPD : EMBD;
    for (int k0 = 0; k0 < K; k0 += 32) {
      __syncthreads();
      { // A tile 128x32, f32 -> bf16 (2 threads per row, 16 floats each)
        const int r = tid >> 1, c0 = (tid & 1) * 16;
        const float* src = A + (size_t)(mb + r) * K + k0 + c0;
        __bf16* d = sA + r * 32 + c0;
#pragma unroll
        for (int c = 0; c < 16; c += 4) {
          float4 f = *(const float4*)(src + c);
          d[c] = (__bf16)f.x; d[c + 1] = (__bf16)f.y;
          d[c + 2] = (__bf16)f.z; d[c + 3] = (__bf16)f.w;
        }
        if (k0 + 32 < K) __builtin_prefetch(src + 32, 0, 3);
      }
      { // B^T tile 256x32: weight row (g*256+n) is contiguous in k
        const int n = tid;
        const float* src = W + (size_t)(g * HYPD + n) * K + k0;
        __bf16* d = sBT + n * 32;
#pragma unroll
        for (int c = 0; c < 32; c += 4) {
          float4 f = *(const float4*)(src + c);
          d[c] = (__bf16)f.x; d[c + 1] = (__bf16)f.y;
          d[c + 2] = (__bf16)f.z; d[c + 3] = (__bf16)f.w;
        }
        if (k0 + 32 < K) __builtin_prefetch(src + 32, 0, 3);
      }
      __syncthreads();
      v16bf a = frag_A(sA + wid * 16 * 32);
#pragma unroll
      for (int t = 0; t < 16; ++t) {
        v16bf b = frag_B(sBT + t * 16 * 32);
        acc[t] = wmma_bf16(a, b, acc[t]);
      }
    }
  }

  // epilogue: + bias, then in-register LayerNorm over the 256 gate columns
#pragma unroll
  for (int t = 0; t < 16; ++t) {
    const float bc = bias[g * HYPD + t * 16 + cl];
#pragma unroll
    for (int v = 0; v < 8; ++v) acc[t][v] += bc;
  }

  float mean[8], rstd[8];
#pragma unroll
  for (int v = 0; v < 8; ++v) {   // row = mb + wid*16 + hl*8 + v
    float s = 0.f, s2 = 0.f;
#pragma unroll
    for (int t = 0; t < 16; ++t) { const float x = acc[t][v]; s += x; s2 += x * x; }
#pragma unroll
    for (int m = 1; m < 16; m <<= 1) {   // reduce across the 16-lane half
      s  += __shfl_xor(s, m, 32);
      s2 += __shfl_xor(s2, m, 32);
    }
    const float mu = s * (1.0f / 256.0f);
    mean[v] = mu;
    rstd[v] = rsqrtf(s2 * (1.0f / 256.0f) - mu * mu + 1e-5f);
  }

#pragma unroll
  for (int t = 0; t < 16; ++t) {
    const int n = t * 16 + cl;
    const float gm = lng[g * HYPD + n], bt = lnb[g * HYPD + n];
#pragma unroll
    for (int v = 0; v < 8; ++v) {
      const int m = mb + wid * 16 + hl * 8 + v;
      ifgo_ln[((size_t)m * 4 + g) * HYPD + n] =
          (acc[t][v] - mean[v]) * rstd[v] * gm + bt;
    }
  }
}

// ------- K2: LSTM pointwise + LN(c) -> h_hat_n (bf16). One block per row. -------
__global__ void __launch_bounds__(256)
k_cell(const float* __restrict__ ifgo, const float* __restrict__ c_hat,
       const float* __restrict__ lncg, const float* __restrict__ lncb,
       __bf16* __restrict__ hout) {
  const int b = blockIdx.x, t = threadIdx.x;
  const float* row = ifgo + (size_t)b * 1024;
  const float ig = row[t], fg = row[HYPD + t], gg = row[2 * HYPD + t], og = row[3 * HYPD + t];
  const float c = sigf(fg) * c_hat[(size_t)b * HYPD + t] + sigf(ig) * tanhf(gg);
  __shared__ float rs[256], rq[256];
  rs[t] = c; rq[t] = c * c;
  __syncthreads();
  for (int s = 128; s > 0; s >>= 1) {
    if (t < s) { rs[t] += rs[t + s]; rq[t] += rq[t + s]; }
    __syncthreads();
  }
  const float mean = rs[0] * (1.0f / 256.0f);
  const float var  = rq[0] * (1.0f / 256.0f) - mean * mean;
  const float cn = (c - mean) * rsqrtf(var + 1e-5f) * lncg[t] + lncb[t];
  hout[(size_t)b * HYPD + t] = (__bf16)(sigf(og) * tanhf(cn));
}

// ------- K3: z projections (zh / zx / zb in one launch via blockIdx.y) -------
__global__ void __launch_bounds__(256)
k_zproj(const __bf16* __restrict__ hn,
        const float* __restrict__ zhW, const float* __restrict__ zhB,
        const float* __restrict__ zxW, const float* __restrict__ zxB,
        const float* __restrict__ zbW,
        __bf16* __restrict__ zh, __bf16* __restrict__ zx,
        float* __restrict__ zb_out) {
  const int mb = blockIdx.x * 64, nblk = blockIdx.y;
  const int proj = (nblk * 128) / HYPD;     // 0=zh, 1=zx, 2=zb
  const int nin0 = (nblk * 128) % HYPD;
  const float* W  = proj == 0 ? zhW : (proj == 1 ? zxW : zbW);
  const float* Bi = proj == 0 ? zhB : (proj == 1 ? zxB : nullptr);
  const int tid = threadIdx.x;
  const int wid = tid >> 5, lane = tid & 31;
  const int wm = wid & 3, wn = wid >> 2;
  const int hl = lane >> 4, cl = lane & 15;

  __shared__ __align__(16) __bf16 sA[64 * 32];
  __shared__ __align__(16) __bf16 sBT[128 * 32];

  v8f acc[4]; zero_acc(acc, 4);

  for (int k0 = 0; k0 < HYPD; k0 += 32) {
    __syncthreads();
    { // A already bf16 in workspace: one 16B vector copy per thread
      const int r = tid >> 2, c0 = (tid & 3) * 8;
      *(uint4*)(sA + r * 32 + c0) =
        *(const uint4*)(hn + (size_t)(mb + r) * HYPD + k0 + c0);
    }
    { // B^T tile 128x32 from linear weight rows (contiguous in k)
      const int n = tid & 127, c0 = (tid >> 7) * 16;
      const float* src = W + (size_t)(nin0 + n) * HYPD + k0 + c0;
      __bf16* d = sBT + n * 32 + c0;
#pragma unroll
      for (int c = 0; c < 16; c += 4) {
        float4 f = *(const float4*)(src + c);
        d[c] = (__bf16)f.x; d[c + 1] = (__bf16)f.y;
        d[c + 2] = (__bf16)f.z; d[c + 3] = (__bf16)f.w;
      }
      if (k0 + 32 < HYPD) __builtin_prefetch(src + 32, 0, 3);
    }
    __syncthreads();
    v16bf a = frag_A(sA + wm * 16 * 32);
#pragma unroll
    for (int t = 0; t < 4; ++t) {
      v16bf b = frag_B(sBT + (wn * 64 + t * 16) * 32);
      acc[t] = wmma_bf16(a, b, acc[t]);
    }
  }

#pragma unroll
  for (int t = 0; t < 4; ++t) {
    const int nin = nin0 + wn * 64 + t * 16 + cl;
    const float bc = Bi ? Bi[nin] : 0.0f;
#pragma unroll
    for (int v = 0; v < 8; ++v) {
      const int m = mb + wm * 16 + hl * 8 + v;
      const float val = acc[t][v] + bc;
      if (proj == 0)      zh[(size_t)m * HYPD + nin] = (__bf16)val;
      else if (proj == 1) zx[(size_t)m * HYPD + nin] = (__bf16)val;
      else // z_b output, transposed to [4][B][NZ] f32
        zb_out[((size_t)(nin >> 6) * BB + m) * NZD + (nin & 63)] = val;
    }
  }
}

// ------- K4: dhw[g] = z[:,g,:] @ Wc[g]  ([8192x64]*[64xN], streamed f32 out) -------
__global__ void __launch_bounds__(256)
k_dhw(const __bf16* __restrict__ z, const __bf16* __restrict__ WcT,
      float* __restrict__ out, int N) {
  const int mb = blockIdx.x * 64, nb = blockIdx.y, g = blockIdx.z;
  const int tid = threadIdx.x;
  const int wid = tid >> 5, lane = tid & 31;
  const int wm = wid & 3, wn = wid >> 2;
  const int hl = lane >> 4, cl = lane & 15;

  __shared__ __align__(16) __bf16 sA[64 * 32];
  __shared__ __align__(16) __bf16 sBT[128 * 32];

  v8f acc[4]; zero_acc(acc, 4);

#pragma unroll
  for (int k0 = 0; k0 < NZD; k0 += 32) {
    __syncthreads();
    { // gate-g slice of z, bf16: one 16B copy per thread
      const int r = tid >> 2, c0 = (tid & 3) * 8;
      *(uint4*)(sA + r * 32 + c0) =
        *(const uint4*)(z + (size_t)(mb + r) * HYPD + g * NZD + k0 + c0);
    }
    { // WcT is N-major bf16: contiguous 32B copies per column
      const int n = tid & 127, w2 = tid >> 7;
      const __bf16* src = WcT + ((size_t)g * N + nb * 128 + n) * NZD + k0 + w2 * 16;
      *(uint4*)(sBT + n * 32 + w2 * 16)     = *(const uint4*)(src);
      *(uint4*)(sBT + n * 32 + w2 * 16 + 8) = *(const uint4*)(src + 8);
    }
    __syncthreads();
    v16bf a = frag_A(sA + wm * 16 * 32);
#pragma unroll
    for (int t = 0; t < 4; ++t) {
      v16bf b = frag_B(sBT + (wn * 64 + t * 16) * 32);
      acc[t] = wmma_bf16(a, b, acc[t]);
    }
  }

  float* og = out + (size_t)g * BB * N;
#pragma unroll
  for (int t = 0; t < 4; ++t) {
    const int ncol = nb * 128 + wn * 64 + t * 16 + cl;
#pragma unroll
    for (int v = 0; v < 8; ++v) {
      const int m = mb + wm * 16 + hl * 8 + v;
      og[(size_t)m * N + ncol] = acc[t][v];
    }
  }
}

// --------------------------------- launcher ---------------------------------
extern "C" void kernel_launch(void* const* d_in, const int* in_sizes, int n_in,
                              void* d_out, int out_size, void* d_ws, size_t ws_size,
                              hipStream_t stream) {
  (void)in_sizes; (void)n_in; (void)out_size; (void)ws_size;
  // inputs (setup_inputs order); x, h, c are unused by the reference outputs
  const float* e        = (const float*)d_in[1];
  const float* h_hat    = (const float*)d_in[4];
  const float* c_hat    = (const float*)d_in[5];
  const float* hyper_Wx = (const float*)d_in[6];
  const float* hyper_Wh = (const float*)d_in[7];
  const float* hyper_b  = (const float*)d_in[8];
  const float* ln_g     = (const float*)d_in[9];
  const float* ln_b     = (const float*)d_in[10];
  const float* lnc_g    = (const float*)d_in[11];
  const float* lnc_b    = (const float*)d_in[12];
  const float* zh_W     = (const float*)d_in[13];
  const float* zh_b     = (const float*)d_in[14];
  const float* zx_W     = (const float*)d_in[15];
  const float* zx_b     = (const float*)d_in[16];
  const float* zb_W     = (const float*)d_in[17];
  const float* dh_W     = (const float*)d_in[18];
  const float* dx_W     = (const float*)d_in[19];
  const float* w_h      = (const float*)d_in[20];
  const float* w_x      = (const float*)d_in[21];

  float* out = (float*)d_out;
  float* out_dhw_h = out;                                  // [4][8192][1024]
  float* out_dhw_x = out + (size_t)4 * BB * HIDD;          // [4][8192][512]
  float* out_zb    = out + (size_t)4 * BB * (HIDD + EMBD); // [4][8192][64]

  // workspace layout (bytes)
  char* w = (char*)d_ws;
  float*  ws_ifgo  = (float*)(w);                                  // 33,554,432
  __bf16* ws_hhatn = (__bf16*)(w + (size_t)33554432);              //  4,194,304
  __bf16* ws_zh    = (__bf16*)(w + (size_t)37748736);              //  4,194,304
  __bf16* ws_zx    = (__bf16*)(w + (size_t)41943040);              //  4,194,304
  __bf16* ws_WcTh  = (__bf16*)(w + (size_t)46137344);              //    524,288
  __bf16* ws_WcTx  = (__bf16*)(w + (size_t)46661632);              //    262,144

  // K0: combined weights Wc[g] = dW[g]^T @ w[g]   (bf16, N-major)
  k_combine<<<dim3(HIDD / 128, 4), 256, 0, stream>>>(dh_W, w_h, ws_WcTh, HIDD);
  k_combine<<<dim3(EMBD / 128, 4), 256, 0, stream>>>(dx_W, w_x, ws_WcTx, EMBD);

  // K1: hyper-cell GEMM + per-gate LN (register-resident LN epilogue)
  k_hyper_ifgo<<<dim3(BB / 128, 4), 256, 0, stream>>>(
      e, h_hat, hyper_Wx, hyper_Wh, hyper_b, ln_g, ln_b, ws_ifgo);

  // K2: LSTM pointwise + LN(c) -> h_hat_n
  k_cell<<<BB, 256, 0, stream>>>(ws_ifgo, c_hat, lnc_g, lnc_b, ws_hhatn);

  // K3: z projections (zh, zx -> ws bf16; zb -> output f32 transposed)
  k_zproj<<<dim3(BB / 64, 6), 256, 0, stream>>>(
      ws_hhatn, zh_W, zh_b, zx_W, zx_b, zb_W, ws_zh, ws_zx, out_zb);

  // K4: dhw_h / dhw_x (the big streamed outputs)
  k_dhw<<<dim3(BB / 64, HIDD / 128, 4), 256, 0, stream>>>(ws_zh, ws_WcTh, out_dhw_h, HIDD);
  k_dhw<<<dim3(BB / 64, EMBD / 128, 4), 256, 0, stream>>>(ws_zx, ws_WcTx, out_dhw_x, EMBD);
}
